// DynamicTransformerEncoderLayer_6708738916880
// MI455X (gfx1250) — matrix operand order repaired and checked by reading, not compiled, not run on hardware
//
#include <hip/hip_runtime.h>

#define TOK   4096   // B*S
#define EMB   768
#define FFN   3072
#define QKV3  2304
#define TDIM  128
#define HIDD  128
#define RRANK 4
#define HEADS 12
#define HDIM  64
#define SEQ   2048
#define BATCH 2

typedef __bf16 bf16_t;
typedef bf16_t v16bf __attribute__((ext_vector_type(16)));
typedef bf16_t v8bf  __attribute__((ext_vector_type(8)));
typedef float  v8f   __attribute__((ext_vector_type(8)));

__device__ __forceinline__ unsigned short f2bf(float f) {
  unsigned int u = __float_as_uint(f);
  u += 0x7fffu + ((u >> 16) & 1u);          // round-to-nearest-even
  return (unsigned short)(u >> 16);
}
__device__ __forceinline__ float bf2f(unsigned short h) {
  return __uint_as_float(((unsigned int)h) << 16);
}

__device__ __forceinline__ v8f wmma_bf16(v16bf a, v16bf b, v8f c) {
  // 8 args: (neg_a, A, neg_b, B, c_mod, C, reuse_a, reuse_b)
  return __builtin_amdgcn_wmma_f32_16x16x32_bf16(false, a, false, b, (short)0, c,
                                                 false, false);
}

// A-operand fragment (16 rows x 32 K). Lane L (0..15): row=row0+L, K = 0..15 contiguous
// Lane L (16..31): row=row0+L-16, K = 16..31. Two 16B loads.
__device__ __forceinline__ v16bf load_frag_a(const unsigned short* base, int ld,
                                             int row0, int k0, int lane) {
  int r  = row0 + (lane & 15);
  int kb = k0 + ((lane >> 4) << 4);
  const unsigned short* rp = base + (size_t)r * ld + kb;
  v8bf lo = *(const v8bf*)(rp);
  v8bf hi = *(const v8bf*)(rp + 8);
  v16bf f;
#pragma unroll
  for (int i = 0; i < 8; ++i) { f[i] = lo[i]; f[i + 8] = hi[i]; }
  return f;
}

// B-operand fragment (32 K x 16 cols), data given row-major as cols x K (NT gemm).
// Lane L (0..15): col=col0+L, K = 0..15 contiguous; Lane 16..31: K = 16..31.
__device__ __forceinline__ v16bf load_frag_b(const unsigned short* base, int ld,
                                             int col0, int k0, int lane) {
  int c  = col0 + (lane & 15);
  int kb = k0 + ((lane >> 4) << 4);
  const unsigned short* rp = base + (size_t)c * ld + kb;
  v8bf lo = *(const v8bf*)(rp);
  v8bf hi = *(const v8bf*)(rp + 8);
  v16bf f;
#pragma unroll
  for (int i = 0; i < 8; ++i) { f[i] = lo[i]; f[i + 8] = hi[i]; }
  return f;
}

// ---------------- elementwise / pack kernels ----------------

__global__ void cast_f32_bf16_k(const float* __restrict__ in,
                                unsigned short* __restrict__ out, int n) {
  int i = blockIdx.x * blockDim.x + threadIdx.x;
  if (i < n) out[i] = f2bf(in[i]);
}

// hA_w2 [F*R, HID] -> W2A' [F, R*HID] : out[f*512 + r*128 + h] = w[(f*4+r)*128 + h]
__global__ void pack_w2a_k(const float* __restrict__ w, unsigned short* __restrict__ out) {
  int i = blockIdx.x * blockDim.x + threadIdx.x;       // < FFN*512
  if (i >= FFN * RRANK * HIDD) return;
  int f = i / (RRANK * HIDD);
  int rem = i - f * (RRANK * HIDD);
  int r = rem / HIDD, h = rem - r * HIDD;
  out[i] = f2bf(w[((size_t)(f * RRANK + r)) * HIDD + h]);
}

// hB_w2 [R*E, HID] -> W2B' [R*HID, E] : out[(r*128+h)*768 + e] = w[(r*768+e)*128 + h]
__global__ void pack_w2b_k(const float* __restrict__ w, unsigned short* __restrict__ out) {
  int i = blockIdx.x * blockDim.x + threadIdx.x;       // < 512*768
  if (i >= RRANK * HIDD * EMB) return;
  int n = i / EMB;
  int e = i - n * EMB;
  int r = n / HIDD, h = n - r * HIDD;
  out[i] = f2bf(w[((size_t)(r * EMB + e)) * HIDD + h]);
}

__global__ void rmsnorm_bf16_k(const float* __restrict__ x, const float* __restrict__ w,
                               unsigned short* __restrict__ out) {
  __shared__ float red[256];
  int row = blockIdx.x;
  const float* xr = x + (size_t)row * EMB;
  float s = 0.f;
  for (int i = threadIdx.x; i < EMB; i += 256) { float v = xr[i]; s += v * v; }
  red[threadIdx.x] = s;
  __syncthreads();
  for (int st = 128; st > 0; st >>= 1) {
    if (threadIdx.x < st) red[threadIdx.x] += red[threadIdx.x + st];
    __syncthreads();
  }
  float scale = rsqrtf(red[0] * (1.0f / EMB) + 1.1920928955078125e-07f);
  for (int i = threadIdx.x; i < EMB; i += 256)
    out[(size_t)row * EMB + i] = f2bf(xr[i] * scale * w[i]);
}

// inter[t,r] = sum_h hB[t,h]*D[t,r*128+h] + sum_e xn2[t,e]*hB_b2[r*768+e]
__global__ void inter_k(const unsigned short* __restrict__ hB, const float* __restrict__ D,
                        const unsigned short* __restrict__ xn2,
                        const float* __restrict__ hB_b2, float* __restrict__ inter) {
  int i = blockIdx.x * blockDim.x + threadIdx.x;       // < TOK*RRANK
  if (i >= TOK * RRANK) return;
  int t = i >> 2, r = i & 3;
  float s = 0.f;
  for (int h = 0; h < HIDD; ++h)
    s += bf2f(hB[(size_t)t * HIDD + h]) * D[(size_t)t * (RRANK * HIDD) + r * HIDD + h];
  for (int e = 0; e < EMB; ++e)
    s += bf2f(xn2[(size_t)t * EMB + e]) * hB_b2[r * EMB + e];
  inter[i] = s;
}

// Cmat[t, r*128+h] = inter[t,r] * hA[t,h]
__global__ void cmat_k(const float* __restrict__ inter, const unsigned short* __restrict__ hA,
                       unsigned short* __restrict__ Cmat) {
  int i = blockIdx.x * blockDim.x + threadIdx.x;       // < TOK*512
  if (i >= TOK * RRANK * HIDD) return;
  int t = i / (RRANK * HIDD);
  int rem = i - t * (RRANK * HIDD);
  int r = rem / HIDD, h = rem - r * HIDD;
  Cmat[i] = f2bf(inter[t * RRANK + r] * bf2f(hA[(size_t)t * HIDD + h]));
}

// h = relu(base + adapt + sum_r inter[t,r]*hA_b2[f*4+r])  -> bf16
__global__ void ffn_combine_k(const float* __restrict__ base, const float* __restrict__ adapt,
                              const float* __restrict__ inter, const float* __restrict__ hA_b2,
                              unsigned short* __restrict__ hbuf) {
  int i = blockIdx.x * blockDim.x + threadIdx.x;       // < TOK*FFN
  if (i >= TOK * FFN) return;
  int t = i / FFN, f = i - t * FFN;
  float s = base[i] + adapt[i];
#pragma unroll
  for (int r = 0; r < 4; ++r) s += inter[t * RRANK + r] * hA_b2[f * RRANK + r];
  hbuf[i] = f2bf(fmaxf(s, 0.f));
}

// ---------------- generic NT GEMM: C[M,N] = A[MxK] x B[NxK]^T (bf16 in, f32 acc) ---
// block = 256 threads (8 waves); block tile = 128 (M) x 64 (N); each wave owns a
// 32x32 register tile = 4 independent WMMA accumulators (2 A-frags x 2 B-frags).
__device__ __forceinline__ void gemm_epilogue(v8f acc, int row0, int col0, int lane,
                                              float* __restrict__ outf,
                                              unsigned short* __restrict__ outbf, int ldc,
                                              const float* __restrict__ addsrc,
                                              const float* __restrict__ bias, int relu) {
  int n = col0 + (lane & 15);
  int rbase = row0 + ((lane >> 4) << 3);
  float bv = bias ? bias[n] : 0.f;
#pragma unroll
  for (int r = 0; r < 8; ++r) {
    int m = rbase + r;
    float v = acc[r] + bv;
    if (addsrc) v += addsrc[(size_t)m * ldc + n];
    if (relu) v = fmaxf(v, 0.f);
    if (outf) outf[(size_t)m * ldc + n] = v;
    if (outbf) outbf[(size_t)m * ldc + n] = f2bf(v);
  }
}

__global__ void gemm_bf16_nt_k(const unsigned short* __restrict__ A, int lda,
                               const unsigned short* __restrict__ Bm, int ldb,
                               int K, float* __restrict__ outf,
                               unsigned short* __restrict__ outbf, int ldc,
                               const float* __restrict__ addsrc,
                               const float* __restrict__ bias, int relu) {
  int lane = threadIdx.x & 31;
  int wave = threadIdx.x >> 5;                          // 0..7 -> 4 (M) x 2 (N)
  int row0 = blockIdx.x * 128 + (wave & 3) * 32;
  int col0 = blockIdx.y * 64 + (wave >> 2) * 32;
  v8f acc00 = {}, acc01 = {}, acc10 = {}, acc11 = {};
  for (int k0 = 0; k0 < K; k0 += 32) {
    v16bf a0 = load_frag_a(A, lda, row0, k0, lane);
    v16bf a1 = load_frag_a(A, lda, row0 + 16, k0, lane);
    v16bf b0 = load_frag_b(Bm, ldb, col0, k0, lane);
    v16bf b1 = load_frag_b(Bm, ldb, col0 + 16, k0, lane);
    acc00 = wmma_bf16(a0, b0, acc00);
    acc01 = wmma_bf16(a0, b1, acc01);
    acc10 = wmma_bf16(a1, b0, acc10);
    acc11 = wmma_bf16(a1, b1, acc11);
  }
  gemm_epilogue(acc00, row0,      col0,      lane, outf, outbf, ldc, addsrc, bias, relu);
  gemm_epilogue(acc01, row0,      col0 + 16, lane, outf, outbf, ldc, addsrc, bias, relu);
  gemm_epilogue(acc10, row0 + 16, col0,      lane, outf, outbf, ldc, addsrc, bias, relu);
  gemm_epilogue(acc11, row0 + 16, col0 + 16, lane, outf, outbf, ldc, addsrc, bias, relu);
}

// ---------------- flash attention (scores^T formulation) ----------------
// grid: (SEQ/64, HEADS, BATCH), block 128 (4 waves, 16 q-rows each).
__global__ void flash_attn_k(const unsigned short* __restrict__ qkv,
                             unsigned short* __restrict__ ctx) {
  __shared__ __align__(16) unsigned short ldsVT[HDIM * 32];  // V^T tile [d][k]
  int lane = threadIdx.x & 31;
  int wave = threadIdx.x >> 5;
  int h = blockIdx.y, b = blockIdx.z;
  int q0 = blockIdx.x * 64 + wave * 16;
  const int ldq = QKV3;
  const unsigned short* Qb = qkv + (size_t)(b * SEQ) * ldq + h * HDIM;
  const unsigned short* Kb = Qb + EMB;
  const unsigned short* Vb = Qb + 2 * EMB;

  // Q^T is the B-operand of scores^T = K x Q^T ; preload both d-steps.
  v16bf qf0 = load_frag_b(Qb, ldq, q0, 0, lane);
  v16bf qf1 = load_frag_b(Qb, ldq, q0, 32, lane);

  float mrun = -1e30f, lrun = 0.f;
  v8f acc0 = {}, acc1 = {}, acc2 = {}, acc3 = {};
  bool hi = (lane >= 16);

  for (int k0 = 0; k0 < SEQ; k0 += 32) {
    __syncthreads();
    for (int i = threadIdx.x; i < 32 * HDIM; i += 128) {
      int kk = i >> 6, d = i & 63;
      ldsVT[d * 32 + kk] = Vb[(size_t)(k0 + kk) * ldq + d];
    }
    __syncthreads();

    // scores^T tiles: s0 keys [k0..k0+15], s1 keys [k0+16..k0+31]; contraction d=64.
    v8f s0 = {}, s1 = {};
    s0 = wmma_bf16(load_frag_a(Kb, ldq, k0, 0, lane), qf0, s0);
    s0 = wmma_bf16(load_frag_a(Kb, ldq, k0, 32, lane), qf1, s0);
    s1 = wmma_bf16(load_frag_a(Kb, ldq, k0 + 16, 0, lane), qf0, s1);
    s1 = wmma_bf16(load_frag_a(Kb, ldq, k0 + 16, 32, lane), qf1, s1);

    float tmax = -1e30f;
#pragma unroll
    for (int r = 0; r < 8; ++r) {
      s0[r] *= 0.125f; s1[r] *= 0.125f;
      tmax = fmaxf(tmax, fmaxf(s0[r], s1[r]));
    }
    tmax = fmaxf(tmax, __shfl_xor(tmax, 16));
    float mnew = fmaxf(mrun, tmax);
    float alpha = __expf(mrun - mnew);

    union { v16bf v; unsigned short u[16]; } pf;
    float lsum = 0.f;
#pragma unroll
    for (int r = 0; r < 8; ++r) {
      float e0 = __expf(s0[r] - mnew);
      float e1 = __expf(s1[r] - mnew);
      lsum += e0 + e1;
      float o0 = __shfl_xor(e0, 16);
      float o1 = __shfl_xor(e1, 16);
      pf.u[r]     = f2bf(hi ? o1 : e0);   // B-frag k run start
      pf.u[r + 8] = f2bf(hi ? e1 : o0);   // B-frag k run tail
    }
    lrun = lrun * alpha + (lsum + __shfl_xor(lsum, 16));
    mrun = mnew;
#pragma unroll
    for (int r = 0; r < 8; ++r) {
      acc0[r] *= alpha; acc1[r] *= alpha; acc2[r] *= alpha; acc3[r] *= alpha;
    }
    // ctx^T += V^T x P^T   (A = V^T from LDS, contraction over 32 keys)
    acc0 = wmma_bf16(load_frag_a(ldsVT, 32, 0, 0, lane), pf.v, acc0);
    acc1 = wmma_bf16(load_frag_a(ldsVT, 32, 16, 0, lane), pf.v, acc1);
    acc2 = wmma_bf16(load_frag_a(ldsVT, 32, 32, 0, lane), pf.v, acc2);
    acc3 = wmma_bf16(load_frag_a(ldsVT, 32, 48, 0, lane), pf.v, acc3);
  }

  float inv = 1.0f / lrun;
  int q = lane & 15;
  int rb = (lane >> 4) << 3;
  unsigned short* op = ctx + (size_t)(b * SEQ + q0 + q) * EMB + h * HDIM;
#pragma unroll
  for (int r = 0; r < 8; ++r) {
    op[rb + r +  0] = f2bf(acc0[r] * inv);
    op[rb + r + 16] = f2bf(acc1[r] * inv);
    op[rb + r + 32] = f2bf(acc2[r] * inv);
    op[rb + r + 48] = f2bf(acc3[r] * inv);
  }
}

// ---------------- host launch ----------------
extern "C" void kernel_launch(void* const* d_in, const int* in_sizes, int n_in,
                              void* d_out, int out_size, void* d_ws, size_t ws_size,
                              hipStream_t stream) {
  const float* src       = (const float*)d_in[0];
  const float* task      = (const float*)d_in[1];
  const float* norm1_w   = (const float*)d_in[2];
  const float* norm2_w   = (const float*)d_in[3];
  const float* in_proj_w = (const float*)d_in[4];
  const float* out_proj_w= (const float*)d_in[5];
  const float* ffn1_w    = (const float*)d_in[6];
  const float* ffn1_b    = (const float*)d_in[7];
  const float* ffn2_w    = (const float*)d_in[8];
  const float* ffn2_b    = (const float*)d_in[9];
  const float* hA_w1     = (const float*)d_in[10];
  const float* hA_b1     = (const float*)d_in[11];
  const float* hA_w2     = (const float*)d_in[12];
  const float* hA_b2     = (const float*)d_in[13];
  const float* hB_w1     = (const float*)d_in[14];
  const float* hB_b1     = (const float*)d_in[15];
  const float* hB_w2     = (const float*)d_in[16];
  const float* hB_b2     = (const float*)d_in[17];
  float* out = (float*)d_out;

  char* p = (char*)d_ws;
  auto carve = [&](size_t bytes) -> void* {
    void* r = (void*)p;
    p += (bytes + 255) & ~(size_t)255;
    return r;
  };
  typedef unsigned short u16;
  u16* wq   = (u16*)carve((size_t)QKV3 * EMB * 2);
  u16* wo   = (u16*)carve((size_t)EMB * EMB * 2);
  u16* w1   = (u16*)carve((size_t)FFN * EMB * 2);
  u16* w2   = (u16*)carve((size_t)EMB * FFN * 2);
  u16* wa1  = (u16*)carve((size_t)HIDD * TDIM * 2);
  u16* wb1  = (u16*)carve((size_t)HIDD * TDIM * 2);
  u16* w2a  = (u16*)carve((size_t)FFN * RRANK * HIDD * 2);
  u16* w2b  = (u16*)carve((size_t)RRANK * HIDD * EMB * 2);
  u16* tbf  = (u16*)carve((size_t)TOK * TDIM * 2);
  u16* xn1  = (u16*)carve((size_t)TOK * EMB * 2);
  u16* qkv  = (u16*)carve((size_t)TOK * QKV3 * 2);
  u16* ctx  = (u16*)carve((size_t)TOK * EMB * 2);
  float* s2 = (float*)carve((size_t)TOK * EMB * 4);
  u16* xn2  = (u16*)carve((size_t)TOK * EMB * 2);
  u16* hA   = (u16*)carve((size_t)TOK * HIDD * 2);
  u16* hB   = (u16*)carve((size_t)TOK * HIDD * 2);
  float* Db = (float*)carve((size_t)TOK * RRANK * HIDD * 4);
  float* it = (float*)carve((size_t)TOK * RRANK * 4);
  u16* Cm   = (u16*)carve((size_t)TOK * RRANK * HIDD * 2);
  float* bs = (float*)carve((size_t)TOK * FFN * 4);
  float* ad = (float*)carve((size_t)TOK * FFN * 4);
  u16* hbuf = (u16*)carve((size_t)TOK * FFN * 2);

  auto cast = [&](const float* in, u16* o, int n) {
    cast_f32_bf16_k<<<dim3((n + 255) / 256), dim3(256), 0, stream>>>(in, o, n);
  };
  cast(in_proj_w, wq, QKV3 * EMB);
  cast(out_proj_w, wo, EMB * EMB);
  cast(ffn1_w, w1, FFN * EMB);
  cast(ffn2_w, w2, EMB * FFN);
  cast(hA_w1, wa1, HIDD * TDIM);
  cast(hB_w1, wb1, HIDD * TDIM);
  cast(task, tbf, TOK * TDIM);
  pack_w2a_k<<<dim3((FFN * RRANK * HIDD + 255) / 256), dim3(256), 0, stream>>>(hA_w2, w2a);
  pack_w2b_k<<<dim3((RRANK * HIDD * EMB + 255) / 256), dim3(256), 0, stream>>>(hB_w2, w2b);

  // attention block
  rmsnorm_bf16_k<<<dim3(TOK), dim3(256), 0, stream>>>(src, norm1_w, xn1);
  gemm_bf16_nt_k<<<dim3(TOK / 128, QKV3 / 64), dim3(256), 0, stream>>>(
      xn1, EMB, wq, EMB, EMB, nullptr, qkv, QKV3, nullptr, nullptr, 0);
  flash_attn_k<<<dim3(SEQ / 64, HEADS, BATCH), dim3(128), 0, stream>>>(qkv, ctx);
  gemm_bf16_nt_k<<<dim3(TOK / 128, EMB / 64), dim3(256), 0, stream>>>(
      ctx, EMB, wo, EMB, EMB, s2, nullptr, EMB, src, nullptr, 0);

  // FFN block with hypernet LoRA
  rmsnorm_bf16_k<<<dim3(TOK), dim3(256), 0, stream>>>(s2, norm2_w, xn2);
  gemm_bf16_nt_k<<<dim3(TOK / 128, HIDD / 64), dim3(256), 0, stream>>>(
      tbf, TDIM, wa1, TDIM, TDIM, nullptr, hA, HIDD, nullptr, hA_b1, 1);
  gemm_bf16_nt_k<<<dim3(TOK / 128, HIDD / 64), dim3(256), 0, stream>>>(
      tbf, TDIM, wb1, TDIM, TDIM, nullptr, hB, HIDD, nullptr, hB_b1, 1);
  gemm_bf16_nt_k<<<dim3(TOK / 128, (RRANK * HIDD) / 64), dim3(256), 0, stream>>>(
      xn2, EMB, w2b, EMB, EMB, Db, nullptr, RRANK * HIDD, nullptr, nullptr, 0);
  inter_k<<<dim3((TOK * RRANK + 255) / 256), dim3(256), 0, stream>>>(hB, Db, xn2, hB_b2, it);
  cmat_k<<<dim3((TOK * RRANK * HIDD + 255) / 256), dim3(256), 0, stream>>>(it, hA, Cm);
  gemm_bf16_nt_k<<<dim3(TOK / 128, FFN / 64), dim3(256), 0, stream>>>(
      xn2, EMB, w1, EMB, EMB, bs, nullptr, FFN, nullptr, ffn1_b, 0);
  gemm_bf16_nt_k<<<dim3(TOK / 128, FFN / 64), dim3(256), 0, stream>>>(
      Cm, RRANK * HIDD, w2a, RRANK * HIDD, RRANK * HIDD, ad, nullptr, FFN, nullptr, nullptr, 0);
  ffn_combine_k<<<dim3((TOK * FFN + 255) / 256), dim3(256), 0, stream>>>(bs, ad, it, hA_b2, hbuf);
  gemm_bf16_nt_k<<<dim3(TOK / 128, EMB / 64), dim3(256), 0, stream>>>(
      hbuf, FFN, w2, FFN, FFN, out, nullptr, EMB, s2, ffn2_b, 0);
}
